// LinearChainCRF_57973468561764
// MI455X (gfx1250) — matrix operand order repaired and checked by reading, not compiled, not run on hardware
//
#include <hip/hip_runtime.h>

typedef __attribute__((ext_vector_type(16))) _Float16 v16h;
typedef __attribute__((ext_vector_type(8)))  _Float16 v8h;
typedef __attribute__((ext_vector_type(8)))  float    v8f;

#define CRF_B 64
#define CRF_T 2048
#define CRF_N 128
#define IMPOSS (-10000.0f)
#define ASTRIDE 136   // halves per expA row: 272B -> bank-spread for b128 loads
#define LOG2E 1.4426950408889634f
#define LN2   0.6931471805599453f

// One block = 8 waves (256 thr) handles 8 batches x 2 channels = 16 sequence rows.
// Wave w owns output state columns [16w, 16w+16). Rows: 0-7 = supervised ch, 8-15 = full ch.
__global__ void __launch_bounds__(256)
crf_fwd(const float* __restrict__ em, const unsigned char* __restrict__ mask,
        const unsigned char* __restrict__ tgt, const float* __restrict__ trans,
        const float* __restrict__ stv_g, const float* __restrict__ env_g,
        const unsigned char* __restrict__ ftr, const unsigned char* __restrict__ fst,
        const unsigned char* __restrict__ fen, float* __restrict__ out)
{
    const int tid  = threadIdx.x;
    const int wave = tid >> 5;
    const int lane = tid & 31;
    const int hi   = lane >> 4;        // half of wave
    const int l16  = lane & 15;
    const int col  = wave * 16 + l16;  // my state column j
    const int bbase = blockIdx.x * 8;

    __shared__ _Float16 expA[16 * ASTRIDE]; // e^{alpha - m}, f16, [row][state]
    __shared__ float maxp[8 * 16];          // per-wave row-max partials [wave][row]
    __shared__ float mfin[16];
    __shared__ float zacc[16];
    __shared__ float zm[16];
    __shared__ int   lenb[8];

    if (tid < 16) { zacc[tid] = 0.0f; zm[tid] = 0.0f; mfin[tid] = -3.0e38f; }

    // ---- lengths: wave w reduces mask row of batch bbase+w (2048 bool bytes) ----
    {
        const uint4* m4 = (const uint4*)(mask + (size_t)(bbase + wave) * CRF_T);
        unsigned acc = 0;
        #pragma unroll
        for (int i = 0; i < 4; ++i) {          // 32 lanes * 4 * 16B = 2048B
            uint4 v = m4[lane * 4 + i];
            acc += v.x + v.y + v.z + v.w;       // per-byte sums <= 16, no carry
        }
        unsigned s = (acc & 255u) + ((acc >> 8) & 255u) + ((acc >> 16) & 255u) + (acc >> 24);
        #pragma unroll
        for (int m = 16; m >= 1; m >>= 1) s += (unsigned)__shfl_xor((int)s, m, 32);
        if (lane == 0) lenb[wave] = (int)s - 1;
    }
    __syncthreads();
    int lenr[8];
    #pragma unroll
    for (int r = 0; r < 8; ++r) lenr[r] = lenb[r];   // batch of row r / r+8 is r

    // ---- constant per-lane vectors: masked start/end transitions for my column ----
    const float stv = fst[col] ? IMPOSS : stv_g[col];
    const float env = fen[col] ? IMPOSS : env_g[col];

    // ---- resident B fragments: exp(trans) for my 16-column tile, K = 0..127 ----
    // layout: frag f, half h -> B[K = 32f + 16*hi + h][col]
    v16h bfr[4];
    #pragma unroll
    for (int f = 0; f < 4; ++f) {
        #pragma unroll
        for (int h = 0; h < 16; ++h) {
            const int k = f * 32 + hi * 16 + h;
            const float tv = ftr[k * CRF_N + col] ? 0.0f : __expf(trans[k * CRF_N + col]);
            bfr[f][h] = (_Float16)tv;
        }
    }

    auto eoff = [&](int r, int t) -> size_t {
        return ((size_t)(bbase + r) * CRF_T + t) * CRF_N + col;
    };

    float al[8];      // my 8 alpha cells (rows hi*8+r, column `col`)
    float mprev[8];   // row shifts from previous step

    // finalize step t: row max (wave shuffle + cross-wave LDS), write expA, commit z
    auto finalize = [&](int t) {
        float lm[8];
        #pragma unroll
        for (int r = 0; r < 8; ++r) {
            float v = al[r];
            v = fmaxf(v, __shfl_xor(v, 1, 16));
            v = fmaxf(v, __shfl_xor(v, 2, 16));
            v = fmaxf(v, __shfl_xor(v, 4, 16));
            v = fmaxf(v, __shfl_xor(v, 8, 16));
            lm[r] = v;
        }
        if (l16 == 0) {
            #pragma unroll
            for (int r = 0; r < 8; ++r) maxp[wave * 16 + hi * 8 + r] = lm[r];
        }
        __syncthreads();
        if (tid < 16) {
            float v = maxp[tid];
            #pragma unroll
            for (int w = 1; w < 8; ++w) v = fmaxf(v, maxp[w * 16 + tid]);
            mfin[tid] = v;
        }
        __syncthreads();
        #pragma unroll
        for (int r = 0; r < 8; ++r) {
            const int row = hi * 8 + r;
            const float m = mfin[row];
            expA[row * ASTRIDE + col] = (_Float16)__builtin_amdgcn_exp2f((al[r] - m) * LOG2E);
            if (t == lenr[r]) {
                atomicAdd(&zacc[row], __builtin_amdgcn_exp2f((al[r] + env - m) * LOG2E));
                if (tid == hi * 16) zm[row] = m;
            }
            mprev[r] = m;
        }
        __syncthreads();
    };

    // ---- t = 0 : alpha0 = masked emissions + start transitions ----
    #pragma unroll
    for (int r = 0; r < 8; ++r) {
        float e = em[eoff(r, 0)];
        if (hi == 0) e = tgt[eoff(r, 0)] ? e : IMPOSS;
        al[r] = e + stv;
    }
    finalize(0);

    // depth-2 emission prefetch pipeline (cold HBM pass: ~2k-cycle latency)
    float em1[8], em2[8]; unsigned char tg1[8], tg2[8];
    #pragma unroll
    for (int r = 0; r < 8; ++r) {
        em1[r] = em[eoff(r, 1)];
        tg1[r] = (hi == 0) ? tgt[eoff(r, 1)] : (unsigned char)1;
        em2[r] = em[eoff(r, 2)];
        tg2[r] = (hi == 0) ? tgt[eoff(r, 2)] : (unsigned char)1;
    }

    // ---- serial recurrence over T ----
    for (int t = 1; t < CRF_T; ++t) {
        float em3[8] = {}; unsigned char tg3[8] = {};
        if (t + 2 < CRF_T) {
            #pragma unroll
            for (int r = 0; r < 8; ++r) {
                em3[r] = em[eoff(r, t + 2)];
                tg3[r] = (hi == 0) ? tgt[eoff(r, t + 2)] : (unsigned char)1;
            }
        }
        if (t + 16 < CRF_T) __builtin_prefetch(em + eoff(0, t + 16), 0, 1);

        // All 4 A fragments loaded up front (distinct regs -> one dscnt wait),
        // then two independent WMMA accumulation chains.
        v16h a[4];
        #pragma unroll
        for (int f = 0; f < 4; ++f) {
            const int base = l16 * ASTRIDE + f * 32 + hi * 16;
            v8h lo = *(const v8h*)&expA[base];
            v8h hs = *(const v8h*)&expA[base + 8];
            #pragma unroll
            for (int i = 0; i < 8; ++i) { a[f][i] = lo[i]; a[f][8 + i] = hs[i]; }
        }
        v8f acc0 = {}, acc1 = {};
        acc0 = __builtin_amdgcn_wmma_f32_16x16x32_f16(false, a[0], false, bfr[0], (short)0, acc0, false, false);
        acc1 = __builtin_amdgcn_wmma_f32_16x16x32_f16(false, a[1], false, bfr[1], (short)0, acc1, false, false);
        acc0 = __builtin_amdgcn_wmma_f32_16x16x32_f16(false, a[2], false, bfr[2], (short)0, acc0, false, false);
        acc1 = __builtin_amdgcn_wmma_f32_16x16x32_f16(false, a[3], false, bfr[3], (short)0, acc1, false, false);

        // alpha' = m_prev + ln(sum) + emission (channel-masked); raw v_log_f32 is log2
        #pragma unroll
        for (int r = 0; r < 8; ++r) {
            float e = em1[r];
            if (hi == 0) e = tg1[r] ? e : IMPOSS;
            const float s = fmaxf(acc0[r] + acc1[r], 1e-30f);
            al[r] = fmaf(LN2, __builtin_amdgcn_logf(s), mprev[r] + e);
        }
        finalize(t);

        #pragma unroll
        for (int r = 0; r < 8; ++r) {
            em1[r] = em2[r]; tg1[r] = tg2[r];
            em2[r] = em3[r]; tg2[r] = tg3[r];
        }
    }

    // ---- loss[b] = -(z_sup - z_full) = z_full - z_sup ----
    if (tid < 8) {
        const float z0 = zm[tid]     + LN2 * __builtin_amdgcn_logf(fmaxf(zacc[tid],     1e-30f));
        const float z1 = zm[tid + 8] + LN2 * __builtin_amdgcn_logf(fmaxf(zacc[tid + 8], 1e-30f));
        out[bbase + tid] = z1 - z0;
    }
}

extern "C" void kernel_launch(void* const* d_in, const int* in_sizes, int n_in,
                              void* d_out, int out_size, void* d_ws, size_t ws_size,
                              hipStream_t stream) {
    (void)in_sizes; (void)n_in; (void)d_ws; (void)ws_size; (void)out_size;
    const float*         emissions = (const float*)d_in[0];
    const unsigned char* mask      = (const unsigned char*)d_in[1];
    const unsigned char* target    = (const unsigned char*)d_in[2];
    const float*         trans     = (const float*)d_in[3];
    const float*         st        = (const float*)d_in[4];
    const float*         en        = (const float*)d_in[5];
    const unsigned char* ftr       = (const unsigned char*)d_in[6];
    const unsigned char* fst       = (const unsigned char*)d_in[7];
    const unsigned char* fen       = (const unsigned char*)d_in[8];
    crf_fwd<<<dim3(CRF_B / 8), dim3(256), 0, stream>>>(
        emissions, mask, target, trans, st, en, ftr, fst, fen, (float*)d_out);
}
